// GGNN_17952963297398
// MI455X (gfx1250) — matrix operand order repaired and checked by reading, not compiled, not run on hardware
//
#include <hip/hip_runtime.h>
#include <hip/hip_bf16.h>
#include <math.h>

// Problem constants (match reference)
#define NN 100000
#define EE 1600000
#define HH 150
#define GG 256
#define LL 5
#define GATEH 200

#define KP 160      // K (=H) padded to multiple of 32 (packed bf16 A stride)
#define MLD 192     // m buffer leading dim (150 -> multiple of 64)
#define AGLD 160    // agg leading dim (== KP so pack is dense)
#define G3LD 512    // gi/gh leading dim (450 -> multiple of 64)
#define GTLD 256    // tanh1 leading dim (200 -> multiple of 64)

typedef __attribute__((ext_vector_type(16))) __bf16 v16bf;
typedef __attribute__((ext_vector_type(8)))  float  v8f;
typedef __attribute__((ext_vector_type(4)))  unsigned int v4u;

union V16BF { v16bf v; v4u q[2]; unsigned short u[16]; };
union V8F   { v8f  v; float f[8]; };

// f32 -> bf16 round-to-nearest-even (bit pattern as u16)
__device__ __forceinline__ unsigned short f2bf(float f) {
    unsigned int u = __float_as_uint(f);
    unsigned int r = u + 0x7FFFu + ((u >> 16) & 1u);
    return (unsigned short)(r >> 16);
}

__device__ __forceinline__ float sigmoidf_(float x) { return 1.0f / (1.0f + __expf(-x)); }

// ---------------------------------------------------------------------------
// Pack f32 [M x Kin] -> bf16 [M x KP], zero-padding K.
// ---------------------------------------------------------------------------
__global__ __launch_bounds__(256)
void pack_a_bf16(const float* __restrict__ in, unsigned short* __restrict__ out,
                 int Kin, int M)
{
    size_t idx = (size_t)blockIdx.x * blockDim.x + threadIdx.x;
    if (idx >= (size_t)M * KP) return;
    int row = (int)(idx / KP);
    int k   = (int)(idx % KP);
    out[idx] = (k < Kin) ? f2bf(in[(size_t)row * Kin + k]) : (unsigned short)0;
}

// ---------------------------------------------------------------------------
// Pack weight into Bt bf16 [Ncp x KP] with Bt[n][k] = Bmat[k][n], zero-padded.
// transB==0: Bmat = B   (B stored [K x Nc], ldb=Nc) -> B[k*ldb+n]
// transB==1: Bmat = B^T (B stored [Nc x K], ldb=K)  -> B[n*ldb+k]
// ---------------------------------------------------------------------------
__global__ __launch_bounds__(256)
void pack_b_bf16(const float* __restrict__ B, unsigned short* __restrict__ Bt,
                 int K, int Nc, int ldb, int transB, int Ncp)
{
    int idx = blockIdx.x * blockDim.x + threadIdx.x;
    if (idx >= Ncp * KP) return;
    int n = idx / KP;
    int k = idx % KP;
    float v = 0.0f;
    if (n < Nc && k < K)
        v = transB ? B[(size_t)n * ldb + k] : B[(size_t)k * ldb + n];
    Bt[idx] = f2bf(v);
}

// ---------------------------------------------------------------------------
// GEMM on packed bf16: C[M x Ncp] = act( Abf[M x KP] @ B (+bias) )
// - A 16-row tile (16 x KP bf16 = 5120 B, contiguous in Abf) is staged into
//   LDS once per block via GLOBAL_LOAD_ASYNC_TO_LDS_B128, then shared by all
//   4 waves through ds_load_b128 (4x less A traffic from L2).
// - Each wave computes a 16x64 output block (4 WMMA accumulators; A reused 4x).
// - M multiple of 16 (100000 ok), Ncp multiple of 64. Hot loop per k-step:
//   2x ds_load_b128 (A) + 8x global b128 (B) + 4x WMMA, branch-free.
// ---------------------------------------------------------------------------
__global__ __launch_bounds__(128)
void gemm_wmma_bf16(const unsigned short* __restrict__ Abf,
                    const unsigned short* __restrict__ Bt,
                    const float* __restrict__ bias, float* __restrict__ C,
                    int Nc, int Ncp, int act)
{
    __shared__ __align__(16) unsigned short Atile[16 * KP];   // 5120 bytes

    const int tid  = threadIdx.x;
    const int wave = tid >> 5;
    const int lane = tid & 31;
    const int row0 = blockIdx.y * 16;
    const int cb0  = (blockIdx.x * 4 + wave) * 64;   // column-block base

    // ---- cooperative async stage of the A tile (before any wave exits) ----
    // 16*KP bf16 = 320 chunks of 16 B; 128 threads -> 3 rounds.
    {
        const unsigned short* gbase = Abf + (size_t)row0 * KP;
        #pragma unroll
        for (int t = 0; t < 3; ++t) {
            int chunk = t * 128 + tid;
            if (chunk < (16 * KP) / 8) {
                unsigned int       loff = (unsigned int)(size_t)&Atile[chunk * 8];
                const unsigned short* g = gbase + chunk * 8;
                asm volatile("global_load_async_to_lds_b128 %0, %1, off"
                             :: "v"(loff), "v"(g) : "memory");
            }
        }
        asm volatile("s_wait_asynccnt 0" ::: "memory");
    }
    __syncthreads();

    if (cb0 >= Ncp) return;                           // wave-uniform

    const int r     = lane & 15;
    const int khalf = lane >> 4;

    // A layout: comps 0..7 -> K = 8*khalf + c ; comps 8..15 -> K = 16+8*khalf+(c-8)
    const unsigned short* ap = &Atile[r * KP + (khalf << 3)];
    // B layout: comps 0..7 -> K = 8*khalf + c ; comps 8..15 -> K = 16+8*khalf+(c-8) (column = Bt row)
    const unsigned short* bp = Bt + (size_t)(cb0 + r) * KP + (khalf << 3);

    V8F acc[4];
    #pragma unroll
    for (int t = 0; t < 4; ++t)
        #pragma unroll
        for (int i = 0; i < 8; ++i) acc[t].f[i] = 0.0f;

    #pragma unroll
    for (int k0 = 0; k0 < KP; k0 += 32) {
        V16BF a;
        a.q[0] = *(const v4u*)(ap + k0);              // ds_load_b128
        a.q[1] = *(const v4u*)(ap + k0 + 16);
        V16BF b[4];
        #pragma unroll
        for (int t = 0; t < 4; ++t) {
            const unsigned short* bt = bp + (size_t)(t << 4) * KP + k0;
            b[t].q[0] = *(const v4u*)(bt);
            b[t].q[1] = *(const v4u*)(bt + 16);
        }
        #pragma unroll
        for (int t = 0; t < 4; ++t)
            acc[t].v = __builtin_amdgcn_wmma_f32_16x16x32_bf16(
                false, a.v, false, b[t].v, (short)0, acc[t].v, false, false);
    }

    #pragma unroll
    for (int t = 0; t < 4; ++t) {
        const int cc = cb0 + (t << 4) + r;
        float bv = 0.0f;
        if (bias && cc < Nc) bv = bias[cc];
        #pragma unroll
        for (int i = 0; i < 8; ++i) {
            float v = acc[t].f[i] + bv;
            if (act == 1) v = tanhf(v);
            C[(size_t)(row0 + i + (khalf << 3)) * Ncp + cc] = v;
        }
    }
}

// ---------------------------------------------------------------------------
// Edge scatter-add: agg[dst[e], :HH] += m[src[e], :HH]  (one wave per edge)
// ---------------------------------------------------------------------------
__global__ __launch_bounds__(256)
void edge_scatter(const float* __restrict__ m, const int* __restrict__ src,
                  const int* __restrict__ dst, float* __restrict__ agg, int E)
{
    int e = blockIdx.x * 8 + (threadIdx.x >> 5);
    if (e >= E) return;
    int lane = threadIdx.x & 31;
    size_t sb = (size_t)src[e] * MLD;
    size_t db = (size_t)dst[e] * AGLD;
    __builtin_prefetch(&m[sb + lane * 4], 0, 0);      // global_prefetch_b8
    #pragma unroll
    for (int j = lane; j < HH; j += 32)
        atomicAdd(&agg[db + j], m[sb + j]);
}

// ---------------------------------------------------------------------------
// GRU elementwise update; gi/gh leading dim G3LD.
// ---------------------------------------------------------------------------
__global__ __launch_bounds__(256)
void gru_elem(const float* __restrict__ gi, const float* __restrict__ gh,
              const float* __restrict__ b_ih, const float* __restrict__ b_hh,
              const float* __restrict__ h, float* __restrict__ hnew)
{
    size_t idx = (size_t)blockIdx.x * blockDim.x + threadIdx.x;
    if (idx >= (size_t)NN * HH) return;
    int node = (int)(idx / HH);
    int j    = (int)(idx % HH);
    const float* gir = gi + (size_t)node * G3LD;
    const float* ghr = gh + (size_t)node * G3LD;
    float ir  = gir[j]          + b_ih[j];
    float iz  = gir[HH + j]     + b_ih[HH + j];
    float inn = gir[2 * HH + j] + b_ih[2 * HH + j];
    float hr  = ghr[j]          + b_hh[j];
    float hz  = ghr[HH + j]     + b_hh[HH + j];
    float hn  = ghr[2 * HH + j] + b_hh[2 * HH + j];
    float rg = sigmoidf_(ir + hr);
    float zg = sigmoidf_(iz + hz);
    float ng = tanhf(inn + rg * hn);
    hnew[idx] = (1.0f - zg) * ng + zg * h[idx];
}

// ---------------------------------------------------------------------------
// gate_raw[n] = tanh1[n,:GATEH] . gw2 + gb2   (tanh1 leading dim GTLD)
// ---------------------------------------------------------------------------
__global__ __launch_bounds__(256)
void gate_dot(const float* __restrict__ t1, const float* __restrict__ gw2,
              const float* __restrict__ gb2, float* __restrict__ gate_raw)
{
    int n = blockIdx.x * 8 + (threadIdx.x >> 5);
    if (n >= NN) return;
    int lane = threadIdx.x & 31;
    float s = 0.0f;
    for (int j = lane; j < GATEH; j += 32)
        s += t1[(size_t)n * GTLD + j] * gw2[j];
    for (int o = 16; o > 0; o >>= 1) s += __shfl_xor(s, o, 32);
    if (lane == 0) gate_raw[n] = s + gb2[0];
}

__global__ __launch_bounds__(256)
void init_seg(float* __restrict__ gmax, float* __restrict__ gsum)
{
    int g = blockIdx.x * blockDim.x + threadIdx.x;
    if (g < GG) { gmax[g] = -INFINITY; gsum[g] = 0.0f; }
}

__device__ __forceinline__ void atomicMaxF(float* addr, float val)
{
    unsigned int* ua = (unsigned int*)addr;
    unsigned int old = *ua;
    while (__uint_as_float(old) < val) {
        unsigned int prev = atomicCAS(ua, old, __float_as_uint(val));
        if (prev == old) break;
        old = prev;
    }
}

__global__ __launch_bounds__(256)
void seg_max(const float* __restrict__ gate_raw, const int* __restrict__ batch,
             float* __restrict__ gmax)
{
    int n = blockIdx.x * blockDim.x + threadIdx.x;
    if (n >= NN) return;
    atomicMaxF(&gmax[batch[n]], gate_raw[n]);
}

__global__ __launch_bounds__(256)
void seg_exp_sum(float* __restrict__ gate_raw, const int* __restrict__ batch,
                 const float* __restrict__ gmax, float* __restrict__ gsum)
{
    int n = blockIdx.x * blockDim.x + threadIdx.x;
    if (n >= NN) return;
    int b = batch[n];
    float e = __expf(gate_raw[n] - gmax[b]);
    gate_raw[n] = e;
    atomicAdd(&gsum[b], e);
}

__global__ __launch_bounds__(256)
void seg_weighted_sum(const float* __restrict__ gate_e, const int* __restrict__ batch,
                      const float* __restrict__ gsum, const float* __restrict__ h,
                      float* __restrict__ out, float* __restrict__ gate_out)
{
    size_t idx = (size_t)blockIdx.x * blockDim.x + threadIdx.x;
    if (idx >= (size_t)NN * HH) return;
    int n = (int)(idx / HH);
    int j = (int)(idx % HH);
    int b = batch[n];
    float w = gate_e[n] / (gsum[b] + 1e-16f);
    if (j == 0) gate_out[n] = w;
    atomicAdd(&out[(size_t)b * HH + j], w * h[idx]);
}

// ---------------------------------------------------------------------------
extern "C" void kernel_launch(void* const* d_in, const int* in_sizes, int n_in,
                              void* d_out, int out_size, void* d_ws, size_t ws_size,
                              hipStream_t stream)
{
    const float* x     = (const float*)d_in[0];
    const int*   eidx  = (const int*)  d_in[1];   // [2, E]
    const int*   batch = (const int*)  d_in[2];   // [N]
    const float* W     = (const float*)d_in[3];   // [L, H, H]
    const float* w_ih  = (const float*)d_in[4];   // [3H, H]
    const float* w_hh  = (const float*)d_in[5];   // [3H, H]
    const float* b_ih  = (const float*)d_in[6];
    const float* b_hh  = (const float*)d_in[7];
    const float* gw1   = (const float*)d_in[8];   // [H, GATE_H]
    const float* gb1   = (const float*)d_in[9];
    const float* gw2   = (const float*)d_in[10];  // [GATE_H, 1]
    const float* gb2   = (const float*)d_in[11];

    const int* esrc = eidx;
    const int* edst = eidx + EE;

    // ---- workspace carve-up -------------------------------------------------
    float* ws = (float*)d_ws;
    const size_t NH = (size_t)NN * HH;
    float* h0       = ws;                              // N*H
    float* h1       = h0 + NH;                         // N*H
    float* mbuf     = h1 + NH;                         // N*MLD
    float* agg      = mbuf + (size_t)NN * MLD;         // N*AGLD
    float* gi       = agg + (size_t)NN * AGLD;         // N*G3LD (tanh1 aliases)
    float* gh       = gi + (size_t)NN * G3LD;          // N*G3LD
    float* gate_raw = gh + (size_t)NN * G3LD;          // N
    float* gmax     = gate_raw + NN;                   // G
    float* gsum     = gmax + GG;                       // G
    unsigned short* h_bf   = (unsigned short*)(gsum + GG);   // N*KP u16
    unsigned short* agg_bf = h_bf + (size_t)NN * KP;         // N*KP u16
    unsigned short* Wt     = agg_bf + (size_t)NN * KP;       // MLD*KP
    unsigned short* wihT   = Wt + MLD * KP;                  // G3LD*KP
    unsigned short* whhT   = wihT + G3LD * KP;               // G3LD*KP
    unsigned short* gw1T   = whhT + G3LD * KP;               // GTLD*KP

    float* out_graphs = (float*)d_out;            // [G, H]
    float* out_gate   = out_graphs + GG * HH;     // [N]

    // h = x
    hipMemcpyAsync(h0, x, NH * sizeof(float), hipMemcpyDeviceToDevice, stream);

    // Pack loop-invariant weights once per launch.
    pack_b_bf16<<<(G3LD * KP + 255) / 256, 256, 0, stream>>>(w_ih, wihT, HH, 3 * HH, HH, 1, G3LD);
    pack_b_bf16<<<(G3LD * KP + 255) / 256, 256, 0, stream>>>(w_hh, whhT, HH, 3 * HH, HH, 1, G3LD);
    pack_b_bf16<<<(GTLD * KP + 255) / 256, 256, 0, stream>>>(gw1, gw1T, HH, GATEH, GATEH, 0, GTLD);

    const int tilesM = NN / 16;                   // 6250, exact
    dim3 blkG(128);

    float* hcur = h0;
    float* hnxt = h1;
    for (int l = 0; l < LL; ++l) {
        // pack activations + this layer's W
        pack_a_bf16<<<(int)(((size_t)NN * KP + 255) / 256), 256, 0, stream>>>(hcur, h_bf, HH, NN);
        pack_b_bf16<<<(MLD * KP + 255) / 256, 256, 0, stream>>>(
            W + (size_t)l * HH * HH, Wt, HH, HH, HH, 0, MLD);

        // m = h @ W[l]    (Ncp = MLD = 192 -> 3 column blocks of 64)
        {
            dim3 grd((MLD / 64 + 3) / 4, tilesM);
            gemm_wmma_bf16<<<grd, blkG, 0, stream>>>(h_bf, Wt, nullptr, mbuf, HH, MLD, 0);
        }
        // agg = scatter_add(m[src] -> dst)
        hipMemsetAsync(agg, 0, (size_t)NN * AGLD * sizeof(float), stream);
        edge_scatter<<<(EE + 7) / 8, 256, 0, stream>>>(mbuf, esrc, edst, agg, EE);
        pack_a_bf16<<<(int)(((size_t)NN * KP + 255) / 256), 256, 0, stream>>>(agg, agg_bf, AGLD, NN);

        // gi = agg @ w_ih^T ; gh = h @ w_hh^T   (Ncp = G3LD = 512 -> 8 blocks)
        {
            dim3 grd((G3LD / 64 + 3) / 4, tilesM);
            gemm_wmma_bf16<<<grd, blkG, 0, stream>>>(agg_bf, wihT, nullptr, gi, 3 * HH, G3LD, 0);
            gemm_wmma_bf16<<<grd, blkG, 0, stream>>>(h_bf, whhT, nullptr, gh, 3 * HH, G3LD, 0);
        }
        // h = GRU(agg, h)
        gru_elem<<<(int)((NH + 255) / 256), 256, 0, stream>>>(gi, gh, b_ih, b_hh, hcur, hnxt);
        float* t = hcur; hcur = hnxt; hnxt = t;
    }

    // Readout: tanh1 = tanh(h @ gw1 + gb1)  [N x GTLD] (reuse gi)
    float* tanh1 = gi;
    pack_a_bf16<<<(int)(((size_t)NN * KP + 255) / 256), 256, 0, stream>>>(hcur, h_bf, HH, NN);
    {
        dim3 grd((GTLD / 64 + 3) / 4, tilesM);
        gemm_wmma_bf16<<<grd, blkG, 0, stream>>>(h_bf, gw1T, gb1, tanh1, GATEH, GTLD, 1);
    }
    gate_dot<<<(NN + 7) / 8, 256, 0, stream>>>(tanh1, gw2, gb2, gate_raw);

    // Segment softmax + weighted sum
    init_seg<<<1, 256, 0, stream>>>(gmax, gsum);
    hipMemsetAsync(out_graphs, 0, (size_t)GG * HH * sizeof(float), stream);
    seg_max<<<(NN + 255) / 256, 256, 0, stream>>>(gate_raw, batch, gmax);
    seg_exp_sum<<<(NN + 255) / 256, 256, 0, stream>>>(gate_raw, batch, gmax, gsum);
    seg_weighted_sum<<<(int)((NH + 255) / 256), 256, 0, stream>>>(
        gate_raw, batch, gsum, hcur, out_graphs, out_gate);
}
